// GATNetHeadsChanged4LayerEmbedding256Dense_31628139168059
// MI455X (gfx1250) — matrix operand order repaired and checked by reading, not compiled, not run on hardware
//
#include <hip/hip_runtime.h>
#include <hip/hip_bf16.h>
#include <cstdint>
#include <cstddef>

// ---------------------------------------------------------------------------
// Types for CDNA5 WMMA (wave32): v_wmma_f32_16x16x32_bf16
// ---------------------------------------------------------------------------
typedef __attribute__((ext_vector_type(16))) __bf16         v16bf;
typedef __attribute__((ext_vector_type(8)))  float          v8f;
typedef __attribute__((ext_vector_type(4)))  unsigned       uv4;
typedef __attribute__((ext_vector_type(4)))  unsigned short usv4;

union Frag { uv4 q[2]; v16bf b; };   // 32 bytes: two b128 loads -> one bf16 fragment

__device__ __forceinline__ unsigned short f2bfbits(float f) {
    unsigned u = __float_as_uint(f);
    u += 0x7FFFu + ((u >> 16) & 1u);   // round-to-nearest-even
    return (unsigned short)(u >> 16);
}
__device__ __forceinline__ float bf2f(unsigned short b) {
    return __uint_as_float((unsigned)b << 16);
}

// order-preserving float <-> uint encoding (atomicMax over signed floats)
__device__ __forceinline__ unsigned f2ord(float f) {
    unsigned u = __float_as_uint(f);
    return (u & 0x80000000u) ? ~u : (u | 0x80000000u);
}
__device__ __forceinline__ float ord2f(unsigned u) {
    u = (u & 0x80000000u) ? (u & 0x7FFFFFFFu) : ~u;
    return __uint_as_float(u);
}

#define N_NODES 16384
#define SLOPE   0.2f

// ---------------------------------------------------------------------------
// f32 -> bf16 conversion (optional fused ReLU), float4 in / 4x bf16 out
// ---------------------------------------------------------------------------
__global__ void cvt_bf16(const float* __restrict__ in, unsigned short* __restrict__ out,
                         int count, int relu)
{
    int i = (int)(blockIdx.x * blockDim.x + threadIdx.x) * 4;
    if (i >= count) return;
    float4 v = *(const float4*)(in + i);
    if (relu) {
        v.x = fmaxf(v.x, 0.f); v.y = fmaxf(v.y, 0.f);
        v.z = fmaxf(v.z, 0.f); v.w = fmaxf(v.w, 0.f);
    }
    usv4 o;
    o[0] = f2bfbits(v.x); o[1] = f2bfbits(v.y);
    o[2] = f2bfbits(v.z); o[3] = f2bfbits(v.w);
    *(usv4*)(out + i) = o;
}

// ---------------------------------------------------------------------------
// Fragment load / WMMA helpers. K compile-time -> the per-tile W offsets
// (t*16*K halves) fold into the global_load IOFFSET immediates: only two
// base pointers (arow, wbase) live in VGPRs.
// ---------------------------------------------------------------------------
template <int NT, int K>
__device__ __forceinline__ void load_step(const unsigned short* __restrict__ arow,
                                          const unsigned short* __restrict__ wbase,
                                          int k, Frag& a, Frag (&b)[NT])
{
    a.q[0] = *(const uv4*)(arow + k);
    a.q[1] = *(const uv4*)(arow + k + 8);
#pragma unroll
    for (int t = 0; t < NT; ++t) {
        b[t].q[0] = *(const uv4*)(wbase + t * 16 * K + k);
        b[t].q[1] = *(const uv4*)(wbase + t * 16 * K + k + 8);
    }
}

template <int NT>
__device__ __forceinline__ void mma_step(const Frag& a, const Frag (&b)[NT], v8f (&acc)[NT])
{
#pragma unroll
    for (int t = 0; t < NT; ++t)
        acc[t] = __builtin_amdgcn_wmma_f32_16x16x32_bf16(
            false, a.b, false, b[t].b, (short)0, acc[t], false, false);
}

// ---------------------------------------------------------------------------
// bf16-WMMA GEMM: C[N,M] = act(A[N,K] @ W[M,K]^T + bias)
// A, W pre-converted to bf16. One wave computes a 16 x (16*NT) strip of C.
// K-loop manually unrolled 2x with two statically-named fragment sets
// (register-resident double buffer, no dynamic indexing). Requires K % 64 == 0.
// Fragment layouts used here (A/B: contiguous k = 16*(lane>>4) + s, same for both):
//   A 16x32: lanes 0-15 K 0..15, lanes 16-31 K 16..31 (rows = lane&15)
//   B 32x16: lanes 0-15 K 0..15, lanes 16-31 K 16..31 (cols = lane&15)
//   C 16x16 f32: lane L, vgpr v -> C[v + 8*(L>>4)][L&15]
// ---------------------------------------------------------------------------
template <int NT, int K, bool OUT_BF16>
__global__ void gemm_wmma(const unsigned short* __restrict__ A,
                          const unsigned short* __restrict__ W,
                          const float* __restrict__ bias,
                          void* __restrict__ Cout,
                          int N, int M, int reluOut)
{
    const int wave   = (int)((blockIdx.x * blockDim.x + threadIdx.x) >> 5);
    const int lane   = (int)(threadIdx.x & 31);
    const int tilesN = M / (16 * NT);
    const int tm     = (wave / tilesN) << 4;
    const int tn0    = (wave % tilesN) * (16 * NT);
    if (tm >= N) return;                       // wave-uniform

    const int r  = lane & 15;
    const int hi = lane >> 4;

    const unsigned short* arow  = A + (size_t)(tm + r) * K + hi * 16;
    const unsigned short* wbase = W + (size_t)(tn0 + r) * K + hi * 16;

    v8f  acc[NT] = {};
    Frag a0, a1;
    Frag b0[NT], b1[NT];

    load_step<NT, K>(arow, wbase, 0, a0, b0);
#pragma unroll 1
    for (int k0 = 0; k0 < K; k0 += 64) {
        load_step<NT, K>(arow, wbase, k0 + 32, a1, b1);     // prefetch odd step
        mma_step<NT>(a0, b0, acc);
        if (k0 + 64 < K)
            load_step<NT, K>(arow, wbase, k0 + 64, a0, b0); // prefetch next even
        mma_step<NT>(a1, b1, acc);
    }

#pragma unroll
    for (int t = 0; t < NT; ++t) {
        const int col = tn0 + t * 16 + r;
        const float bv = bias ? bias[col] : 0.f;
#pragma unroll
        for (int v = 0; v < 8; ++v) {
            const int row = tm + v + hi * 8;
            float o = acc[t][v] + bv;
            if (reluOut) o = fmaxf(o, 0.f);
            if (OUT_BF16)
                ((unsigned short*)Cout)[(size_t)row * M + col] = f2bfbits(o);
            else
                ((float*)Cout)[(size_t)row * M + col] = o;
        }
    }
}

// ---------------------------------------------------------------------------
// Attention logits: one wave per (node, head); 256-wide dot products
// ---------------------------------------------------------------------------
__global__ void att_reduce(const float* __restrict__ h,
                           const float* __restrict__ att_src,
                           const float* __restrict__ att_dst,
                           float* __restrict__ a_src, float* __restrict__ a_dst)
{
    const int wid  = (int)((blockIdx.x * blockDim.x + threadIdx.x) >> 5);
    const int lane = (int)(threadIdx.x & 31);
    if (wid >= N_NODES * 2) return;
    const int node = wid >> 1, head = wid & 1;
    const float* hp = h + (size_t)node * 512 + head * 256;
    const float* as = att_src + head * 256;
    const float* ad = att_dst + head * 256;
    float s1 = 0.f, s2 = 0.f;
    for (int c = lane; c < 256; c += 32) {
        const float v = hp[c];
        s1 += v * as[c]; s2 += v * ad[c];
    }
#pragma unroll
    for (int off = 16; off; off >>= 1) {
        s1 += __shfl_down(s1, off, 32);
        s2 += __shfl_down(s2, off, 32);
    }
    if (lane == 0) { a_src[wid] = s1; a_dst[wid] = s2; }
}

// ---------------------------------------------------------------------------
// Init: agg <- broadcast(conv_b), m <- enc(-inf), s <- 0
// ---------------------------------------------------------------------------
__global__ void init_bufs(float* __restrict__ agg, const float* __restrict__ conv_b,
                          unsigned* __restrict__ m_ord, float* __restrict__ s_sum)
{
    const size_t i = blockIdx.x * (size_t)blockDim.x + threadIdx.x;
    if (i < (size_t)N_NODES * 512) agg[i] = conv_b[i & 511];
    if (i < (size_t)N_NODES * 2) { m_ord[i] = f2ord(-INFINITY); s_sum[i] = 0.f; }
}

// ---------------------------------------------------------------------------
// Pass 1: edge logits (leaky-relu) + segment-max via ordered-uint atomicMax
// ---------------------------------------------------------------------------
__global__ void edge_logits(const long long* __restrict__ ei,
                            const float* __restrict__ a_src,
                            const float* __restrict__ a_dst,
                            float* __restrict__ e_edge,
                            unsigned* __restrict__ m_ord, int E, int Etot)
{
    const int idx = (int)(blockIdx.x * blockDim.x + threadIdx.x);
    if (idx >= Etot * 2) return;
    const int e = idx >> 1, hd = idx & 1;
    long long s, d;
    if (e < E) { s = ei[e]; d = ei[E + e]; } else { s = d = (long long)(e - E); }
    float v = a_src[s * 2 + hd] + a_dst[d * 2 + hd];
    v = (v > 0.f) ? v : SLOPE * v;
    e_edge[idx] = v;
    atomicMax(m_ord + (size_t)d * 2 + hd, f2ord(v));
}

// ---------------------------------------------------------------------------
// Pass 2: exp(e - max) + segment-sum
// ---------------------------------------------------------------------------
__global__ void edge_exp(const long long* __restrict__ ei,
                         float* __restrict__ e_edge,
                         const unsigned* __restrict__ m_ord,
                         float* __restrict__ s_sum, int E, int Etot)
{
    const int idx = (int)(blockIdx.x * blockDim.x + threadIdx.x);
    if (idx >= Etot * 2) return;
    const int e = idx >> 1, hd = idx & 1;
    long long d;
    if (e < E) { d = ei[E + e]; } else { d = (long long)(e - E); }
    const float mm = ord2f(m_ord[(size_t)d * 2 + hd]);
    const float w = __expf(e_edge[idx] - mm);
    e_edge[idx] = w;
    atomicAdd(s_sum + (size_t)d * 2 + hd, w);
}

// ---------------------------------------------------------------------------
// Pass 3: weighted scatter-add: agg[dst] += alpha * h[src] (512 ch, float4)
// One 128-thread block per edge; accumulator is L2-resident (32 MB).
// ---------------------------------------------------------------------------
__global__ void edge_agg(const long long* __restrict__ ei,
                         const float* __restrict__ e_edge,
                         const float* __restrict__ s_sum,
                         const float* __restrict__ h,
                         float* __restrict__ agg, int E)
{
    const int e = (int)blockIdx.x;
    const int c = (int)threadIdx.x * 4;          // 128 threads * 4 = 512 channels
    long long s, d;
    if (e < E) { s = ei[e]; d = ei[E + e]; } else { s = d = (long long)(e - E); }
    const int hd = c >> 8;
    const float alpha = e_edge[(size_t)e * 2 + hd] / s_sum[(size_t)d * 2 + hd];
    const float4 hv = *(const float4*)(h + (size_t)s * 512 + c);
    float* out = agg + (size_t)d * 512 + c;
    atomicAdd(out + 0, alpha * hv.x);
    atomicAdd(out + 1, alpha * hv.y);
    atomicAdd(out + 2, alpha * hv.z);
    atomicAdd(out + 3, alpha * hv.w);
}

// ---------------------------------------------------------------------------
// Final 32->3 linear (bf16 activations in) + squared-norm
// ---------------------------------------------------------------------------
__global__ void final_lin(const unsigned short* __restrict__ h5b,
                          const float* __restrict__ W4, const float* __restrict__ b4,
                          float* __restrict__ p, float* __restrict__ sq)
{
    const int n = (int)(blockIdx.x * blockDim.x + threadIdx.x);
    if (n >= N_NODES) return;
    const unsigned short* hp = h5b + (size_t)n * 32;
    float o[3];
#pragma unroll
    for (int r = 0; r < 3; ++r) {
        float acc = b4[r];
        const float* w = W4 + r * 32;
#pragma unroll
        for (int k = 0; k < 32; ++k) acc += bf2f(hp[k]) * w[k];
        o[r] = acc;
    }
    p[n * 3 + 0] = o[0]; p[n * 3 + 1] = o[1]; p[n * 3 + 2] = o[2];
    sq[n] = o[0] * o[0] + o[1] * o[1] + o[2] * o[2];
}

// ---------------------------------------------------------------------------
// safe_cdist: D[i,j] = sqrt(max(|pi|^2+|pj|^2-2 pi.pj, 0)); float4 stores.
// 1 GiB output -> store-bandwidth bound (~46 us at 23.3 TB/s).
// ---------------------------------------------------------------------------
__global__ void cdist_k(const float* __restrict__ p, const float* __restrict__ sq,
                        float* __restrict__ D)
{
    const int i  = (int)blockIdx.y;
    const int j0 = (int)(blockIdx.x * blockDim.x + threadIdx.x) * 4;
    const float xi = p[i * 3 + 0], yi = p[i * 3 + 1], zi = p[i * 3 + 2];
    const float si = sq[i];
    float4 out;
    float* o = (float*)&out;
#pragma unroll
    for (int u = 0; u < 4; ++u) {
        const int j = j0 + u;
        const float dot = xi * p[j * 3 + 0] + yi * p[j * 3 + 1] + zi * p[j * 3 + 2];
        float d2 = si + sq[j] - 2.f * dot;
        d2 = fmaxf(d2, 0.f);
        o[u] = (d2 > 0.f) ? sqrtf(d2) : 0.f;
    }
    *(float4*)(D + (size_t)i * N_NODES + j0) = out;
}

// ---------------------------------------------------------------------------
// Orchestration
// ---------------------------------------------------------------------------
extern "C" void kernel_launch(void* const* d_in, const int* in_sizes, int n_in,
                              void* d_out, int out_size, void* d_ws, size_t ws_size,
                              hipStream_t stream)
{
    const float*     x       = (const float*)d_in[0];
    const long long* ei      = (const long long*)d_in[1];
    const float*     conv_W  = (const float*)d_in[2];
    const float*     att_src = (const float*)d_in[3];
    const float*     att_dst = (const float*)d_in[4];
    const float*     conv_b  = (const float*)d_in[5];
    const float*     Wa = (const float*)d_in[6];  const float* ba = (const float*)d_in[7];
    const float*     W1 = (const float*)d_in[8];  const float* b1 = (const float*)d_in[9];
    const float*     W2 = (const float*)d_in[10]; const float* b2 = (const float*)d_in[11];
    const float*     W3 = (const float*)d_in[12]; const float* b3 = (const float*)d_in[13];
    const float*     W4 = (const float*)d_in[14]; const float* b4 = (const float*)d_in[15];
    float* D = (float*)d_out;

    const int N = N_NODES;
    const int E = in_sizes[1] / 2;       // 524288
    const int Etot = E + N;              // + self loops

    // workspace layout (bytes)
    char* base = (char*)d_ws;
    float*          h      = (float*)(base);                        // [N,512] f32 32 MB
    float*          agg    = (float*)(base +  33554432ull);         // [N,512] f32 32 MB
    unsigned short* aggb   = (unsigned short*)(base + 67108864ull); // [N,512] bf16 16 MB
    unsigned short* xb     = (unsigned short*)(base + 83886080ull); // [N,256] bf16 8 MB
    unsigned short* h2b    = (unsigned short*)(base + 92274688ull); // [N,256] bf16 8 MB
    float*          e_edge = (float*)(base + 100663296ull);         // [Etot,2] 4.33 MB
    float*          a_srcb = (float*)(base + 104988672ull);         // [N,2]
    float*          a_dstb = (float*)(base + 105119744ull);         // [N,2]
    unsigned*       m_ord  = (unsigned*)(base + 105250816ull);      // [N,2]
    float*          s_sum  = (float*)(base + 105381888ull);         // [N,2]
    float*          p      = (float*)(base + 105512960ull);         // [N,3]
    float*          sq     = (float*)(base + 105709568ull);         // [N]
    unsigned short* cWb    = (unsigned short*)(base + 105775104ull);// 512x256 bf16
    unsigned short* Wab    = (unsigned short*)(base + 106037248ull);// 256x512 bf16
    unsigned short* W1b    = (unsigned short*)(base + 106299392ull);// 128x256 bf16
    unsigned short* W2b    = (unsigned short*)(base + 106364928ull);// 64x128 bf16
    unsigned short* W3b    = (unsigned short*)(base + 106381312ull);// 32x64 bf16
    // xb is dead after GEMM1 -> alias late bf16 activations into it
    unsigned short* h3b = xb;                                       // [N,128] 4 MB
    unsigned short* h4b = (unsigned short*)((char*)xb + 4194304ull);// [N,64]  2 MB
    unsigned short* h5b = (unsigned short*)((char*)xb + 6291456ull);// [N,32]  1 MB
    (void)n_in; (void)out_size; (void)ws_size;

    auto cvtB = [](int count) { return (count / 4 + 255) / 256; };
    // waves = (N/16) * (M/(16*NT)); blocks = waves*32/256
    auto gB = [N](int M, int nt) { return ((N >> 4) * (M / (16 * nt)) * 32) / 256; };

    // 0. one-time bf16 conversions (weights + x)
    cvt_bf16<<<cvtB(N * 256), 256, 0, stream>>>(x, xb, N * 256, 0);
    cvt_bf16<<<cvtB(512 * 256), 256, 0, stream>>>(conv_W, cWb, 512 * 256, 0);
    cvt_bf16<<<cvtB(256 * 512), 256, 0, stream>>>(Wa, Wab, 256 * 512, 0);
    cvt_bf16<<<cvtB(128 * 256), 256, 0, stream>>>(W1, W1b, 128 * 256, 0);
    cvt_bf16<<<cvtB(64 * 128), 256, 0, stream>>>(W2, W2b, 64 * 128, 0);
    cvt_bf16<<<cvtB(32 * 64), 256, 0, stream>>>(W3, W3b, 32 * 64, 0);

    // 1. h = x @ conv_W^T (f32 out: needed by attention + edge gather)
    gemm_wmma<4, 256, false><<<gB(512, 4), 256, 0, stream>>>(xb, cWb, nullptr, h, N, 512, 0);
    // 2. attention logits
    att_reduce<<<(N * 2 * 32) / 256, 256, 0, stream>>>(h, att_src, att_dst, a_srcb, a_dstb);
    // 3. init accumulators (agg <- bias broadcast; m <- -inf; s <- 0)
    init_bufs<<<(N * 512) / 256, 256, 0, stream>>>(agg, conv_b, m_ord, s_sum);
    // 4-6. edge softmax + aggregation
    const int eth = Etot * 2;
    edge_logits<<<(eth + 255) / 256, 256, 0, stream>>>(ei, a_srcb, a_dstb, e_edge, m_ord, E, Etot);
    edge_exp<<<(eth + 255) / 256, 256, 0, stream>>>(ei, e_edge, m_ord, s_sum, E, Etot);
    edge_agg<<<Etot, 128, 0, stream>>>(ei, e_edge, s_sum, h, agg, E);
    // 7. GAT ReLU folded into the conversion; MLP GEMMs store bf16 directly
    cvt_bf16<<<cvtB(N * 512), 256, 0, stream>>>(agg, aggb, N * 512, 1);
    gemm_wmma<4, 512, true><<<gB(256, 4), 256, 0, stream>>>(aggb, Wab, ba, h2b, N, 256, 1);
    gemm_wmma<4, 256, true><<<gB(128, 4), 256, 0, stream>>>(h2b, W1b, b1, h3b, N, 128, 1);
    gemm_wmma<4, 128, true><<<gB(64, 4), 256, 0, stream>>>(h3b, W2b, b2, h4b, N, 64, 1);
    gemm_wmma<2, 64, true><<<gB(32, 2), 256, 0, stream>>>(h4b, W3b, b3, h5b, N, 32, 1);
    // 11. 32->3 projection + norms
    final_lin<<<N / 256, 256, 0, stream>>>(h5b, W4, b4, p, sq);
    // 12. 16384x16384 distance matrix (1 GiB, store-bound)
    dim3 cg(N / 1024, N);
    cdist_k<<<cg, 256, 0, stream>>>(p, sq, D);
}
